// LoRAAttention_70609262346598
// MI455X (gfx1250) — matrix operand order repaired and checked by reading, not compiled, not run on hardware
//
#include <hip/hip_runtime.h>
#include <hip/hip_bf16.h>
#include <stdint.h>

#define D_MODEL   2048
#define S_LEN     2048
#define BATCH     4
#define N_HEADS   16
#define HEAD_DIM  128
#define R_LORA    8
#define LORA_SCALE 2.0f
#define M_ROWS    (BATCH * S_LEN)   // 8192

typedef __bf16 bf16;
typedef __attribute__((ext_vector_type(4)))  __bf16 bf16x4;
typedef __attribute__((ext_vector_type(8)))  __bf16 bf16x8;
typedef __attribute__((ext_vector_type(16))) __bf16 bf16x16;
typedef __attribute__((ext_vector_type(8)))  float  floatx8;

// ---------- CDNA5 async global->LDS DMA (ASYNCcnt-tracked) ----------
#if defined(__has_builtin)
#  if __has_builtin(__builtin_amdgcn_global_load_async_to_lds_b128)
#    define HAVE_ASYNC_LDS 1
#  endif
#endif
#ifndef HAVE_ASYNC_LDS
#  define HAVE_ASYNC_LDS 0
#endif

#if HAVE_ASYNC_LDS
// Builtin prototype (from compiler diagnostic): pointee is GCC-vector int4,
// global source in AS(1), LDS destination in AS(3), imm offset, imm cpol.
typedef int v4i __attribute__((vector_size(4 * sizeof(int))));
typedef v4i __attribute__((address_space(1)))* gas_v4i_ptr;
typedef v4i __attribute__((address_space(3)))* lds_v4i_ptr;

static __device__ __forceinline__ void async_ld_b128(const void* g, void* l) {
  // Integer round-trips: global addr numeric value is the AS1 address;
  // LDS aperture: flat_addr[31:0] == LDS byte offset.
  __builtin_amdgcn_global_load_async_to_lds_b128(
      (gas_v4i_ptr)(uintptr_t)g,
      (lds_v4i_ptr)(uint32_t)(uintptr_t)l,
      /*offset=*/0, /*cpol=*/0);
}
static __device__ __forceinline__ void wait_async_all() {
#if __has_builtin(__builtin_amdgcn_s_wait_asynccnt)
  __builtin_amdgcn_s_wait_asynccnt(0);
#else
  asm volatile("s_wait_asynccnt 0" ::: "memory");
#endif
}
#endif

// ---------- small helpers ----------
static __device__ __forceinline__ bf16 f2bf(float f) {
  uint32_t u = __builtin_bit_cast(uint32_t, f);
  u += 0x7FFFu + ((u >> 16) & 1u);              // round-to-nearest-even
  uint16_t h = (uint16_t)(u >> 16);
  return __builtin_bit_cast(bf16, h);
}
static __device__ __forceinline__ float bf2f(bf16 b) {
  uint16_t h = __builtin_bit_cast(uint16_t, b);
  uint32_t u = ((uint32_t)h) << 16;
  return __builtin_bit_cast(float, u);
}
static __device__ __forceinline__ float toF(float v) { return v; }
static __device__ __forceinline__ float toF(bf16 v)  { return bf2f(v); }

static __device__ __forceinline__ bf16x8 ld8(const bf16* p) {
  return *(const bf16x8*)p;
}
static __device__ __forceinline__ bf16x16 mk16(bf16x8 lo, bf16x8 hi) {
  bf16x16 r;
#pragma unroll
  for (int i = 0; i < 8; i++) { r[i] = lo[i]; r[i + 8] = hi[i]; }
  return r;
}
static __device__ __forceinline__ floatx8 fzero8() {
  floatx8 z;
#pragma unroll
  for (int i = 0; i < 8; i++) z[i] = 0.0f;
  return z;
}
static __device__ __forceinline__ floatx8 wmma_bf16(bf16x16 a, bf16x16 b, floatx8 c) {
  // D = A(16x32 bf16) * B(32x16 bf16) + C(16x16 f32)
  return __builtin_amdgcn_wmma_f32_16x16x32_bf16(false, a, false, b, (short)0, c,
                                                 false, false);
}
static __device__ __forceinline__ void storeOut(float* p, float v) { *p = v; }
static __device__ __forceinline__ void storeOut(bf16* p, float v)  { *p = f2bf(v); }

// ---------- 1) f32 -> bf16 cast ----------
__global__ __launch_bounds__(256) void cast_f32_bf16_kernel(const float* __restrict__ src,
                                                            bf16* __restrict__ dst, int n4) {
  int i = blockIdx.x * 256 + threadIdx.x;
  if (i < n4) {
    float4 v = ((const float4*)src)[i];
    bf16x4 o;
    o[0] = f2bf(v.x); o[1] = f2bf(v.y); o[2] = f2bf(v.z); o[3] = f2bf(v.w);
    ((bf16x4*)dst)[i] = o;
  }
}

// ---------- 2) P = (X @ A^T) * scale  -> [M, 8] ----------
template <typename InT>
__global__ __launch_bounds__(256) void lora_p_kernel(const InT* __restrict__ X,
                                                     const float* __restrict__ A,
                                                     float* __restrict__ Pout, int Kdim) {
  int row = blockIdx.x;
  int tid = threadIdx.x;
  float acc[R_LORA];
#pragma unroll
  for (int r = 0; r < R_LORA; r++) acc[r] = 0.0f;
  const InT* xr = X + (size_t)row * Kdim;
  for (int d = tid; d < Kdim; d += 256) {
    float xv = toF(xr[d]);
#pragma unroll
    for (int r = 0; r < R_LORA; r++) acc[r] += xv * A[r * Kdim + d];
  }
  __shared__ float red[R_LORA][256];
#pragma unroll
  for (int r = 0; r < R_LORA; r++) red[r][tid] = acc[r];
  __syncthreads();
  for (int s = 128; s > 0; s >>= 1) {
    if (tid < s) {
#pragma unroll
      for (int r = 0; r < R_LORA; r++) red[r][tid] += red[r][tid + s];
    }
    __syncthreads();
  }
  if (tid < R_LORA) Pout[(size_t)row * R_LORA + tid] = red[tid][0] * LORA_SCALE;
}

// ---------- 3) Y = X @ W^T + P @ BL^T   (WMMA bf16, f32 accum) ----------
// X: [M,K] bf16 row-major; W: [N,K] bf16 row-major; P: [M,8] f32; BL: [N,8] f32
// Block tile 128x128, 8 waves (2 M x 4 N), wave tile 64x32, K step 32.
// Async path: double-buffered LDS, DMA of tile k+1 overlaps WMMA of tile k,
// one barrier per K-step.
template <typename OutT>
__global__ __launch_bounds__(256) void gemm_lora_kernel(const bf16* __restrict__ X,
                                                        const bf16* __restrict__ W,
                                                        const float* __restrict__ P,
                                                        const float* __restrict__ BL,
                                                        OutT* __restrict__ Y,
                                                        int Ndim, int Kdim) {
#if HAVE_ASYNC_LDS
  __shared__ __attribute__((aligned(16))) bf16 Al[2][128][40];
  __shared__ __attribute__((aligned(16))) bf16 Bl[2][128][40];
#else
  __shared__ __attribute__((aligned(16))) bf16 Al[1][128][40];
  __shared__ __attribute__((aligned(16))) bf16 Bl[1][128][40];
#endif
  const int tid = threadIdx.x, lane = tid & 31, wave = tid >> 5;
  const int waveM = wave >> 2, waveN = wave & 3;
  const int m0 = blockIdx.y * 128, n0 = blockIdx.x * 128;
  const int mrow = lane & 15;
  const int khA = (lane < 16) ? 0 : 8;    // A-frag K half
  const int kbB = (lane < 16) ? 0 : 16;   // B-frag K base
  const int srow = tid >> 1, sch = (tid & 1) * 16;  // staging: 2 thr/row, 16 elems each

  floatx8 acc[4][2];
#pragma unroll
  for (int i = 0; i < 4; i++)
#pragma unroll
    for (int j = 0; j < 2; j++) acc[i][j] = fzero8();

#if HAVE_ASYNC_LDS
  // prologue: DMA first K-tile into buffer 0 (4 async ops / thread)
  {
    const bf16* xs = X + (size_t)(m0 + srow) * Kdim + sch;
    const bf16* ws = W + (size_t)(n0 + srow) * Kdim + sch;
    async_ld_b128(xs,     &Al[0][srow][sch]);
    async_ld_b128(xs + 8, &Al[0][srow][sch + 8]);
    async_ld_b128(ws,     &Bl[0][srow][sch]);
    async_ld_b128(ws + 8, &Bl[0][srow][sch + 8]);
  }
#endif

  for (int k0 = 0; k0 < Kdim; k0 += 32) {
#if HAVE_ASYNC_LDS
    const int cur = (k0 >> 5) & 1;
    wait_async_all();        // my DMA into Al/Bl[cur] (and older) complete
    __syncthreads();         // everyone's DMA landed; everyone done reading buf cur^1
    if (k0 + 32 < Kdim) {    // overlap: DMA next tile while we run WMMAs on cur
      const bf16* xs = X + (size_t)(m0 + srow) * Kdim + (k0 + 32) + sch;
      const bf16* ws = W + (size_t)(n0 + srow) * Kdim + (k0 + 32) + sch;
      async_ld_b128(xs,     &Al[cur ^ 1][srow][sch]);
      async_ld_b128(xs + 8, &Al[cur ^ 1][srow][sch + 8]);
      async_ld_b128(ws,     &Bl[cur ^ 1][srow][sch]);
      async_ld_b128(ws + 8, &Bl[cur ^ 1][srow][sch + 8]);
    }
#else
    const int cur = 0;
#pragma unroll
    for (int i = 0; i < 2; i++) {
      int c = tid + 256 * i;              // 512 16B chunks per tile
      int row = c >> 2, ch = (c & 3) * 8;
      *(bf16x8*)&Al[0][row][ch] = ld8(X + (size_t)(m0 + row) * Kdim + k0 + ch);
      *(bf16x8*)&Bl[0][row][ch] = ld8(W + (size_t)(n0 + row) * Kdim + k0 + ch);
    }
    if (k0 + 32 < Kdim) {
      __builtin_prefetch(X + (size_t)(m0 + (tid >> 1)) * Kdim + k0 + 32, 0, 1);
      __builtin_prefetch(W + (size_t)(n0 + (tid >> 1)) * Kdim + k0 + 32, 0, 1);
    }
    __syncthreads();
#endif

    bf16x16 af[4], bfr[2];
#pragma unroll
    for (int ti = 0; ti < 4; ti++) {
      int r = waveM * 64 + ti * 16 + mrow;
      af[ti] = mk16(ld8(&Al[cur][r][khA]), ld8(&Al[cur][r][khA + 16]));
    }
#pragma unroll
    for (int tj = 0; tj < 2; tj++) {
      int c = waveN * 32 + tj * 16 + mrow;
      bfr[tj] = mk16(ld8(&Bl[cur][c][khA]), ld8(&Bl[cur][c][khA + 16]));
    }
#pragma unroll
    for (int ti = 0; ti < 4; ti++)
#pragma unroll
      for (int tj = 0; tj < 2; tj++)
        acc[ti][tj] = wmma_bf16(af[ti], bfr[tj], acc[ti][tj]);

#if !HAVE_ASYNC_LDS
    __syncthreads();
#endif
  }

  // Epilogue: + rank-8 LoRA, convert & store
  const int rhalf = (lane >> 4) << 3;
#pragma unroll
  for (int tj = 0; tj < 2; tj++) {
    int gcol = n0 + waveN * 32 + tj * 16 + (lane & 15);
    float4 bv0 = *(const float4*)(BL + (size_t)gcol * 8);
    float4 bv1 = *(const float4*)(BL + (size_t)gcol * 8 + 4);
#pragma unroll
    for (int ti = 0; ti < 4; ti++) {
      int growb = m0 + waveM * 64 + ti * 16 + rhalf;
#pragma unroll
      for (int j = 0; j < 8; j++) {
        float4 pv0 = *(const float4*)(P + (size_t)(growb + j) * 8);
        float4 pv1 = *(const float4*)(P + (size_t)(growb + j) * 8 + 4);
        float lora = pv0.x * bv0.x + pv0.y * bv0.y + pv0.z * bv0.z + pv0.w * bv0.w +
                     pv1.x * bv1.x + pv1.y * bv1.y + pv1.z * bv1.z + pv1.w * bv1.w;
        storeOut(Y + (size_t)(growb + j) * Ndim + gcol, acc[ti][tj][j] + lora);
      }
    }
  }
}

// ---------- 4) causal flash attention ----------
// qkv: [B, S, 3*D] bf16 (q at 0, k at +D, v at +2D, each [H,Dh] per row)
// out: [B, S, D]   bf16, head-concatenated (matches reference transpose/reshape)
__global__ __launch_bounds__(128) void attn_kernel(const bf16* __restrict__ qkv,
                                                   bf16* __restrict__ out) {
  __shared__ __attribute__((aligned(16))) bf16 Kl[32][136];      // [k][dh]
  __shared__ __attribute__((aligned(16))) bf16 Vt[128][40];      // [dh][k] (transposed)
  __shared__ __attribute__((aligned(16))) bf16 Pl[4][16][40];    // per-wave P tile

  const int tid = threadIdx.x, lane = tid & 31, wave = tid >> 5;
  const int bh = blockIdx.y;
  const int b = bh >> 4, h = bh & 15;
  const int q0 = blockIdx.x * 64;
  const int qbase = q0 + wave * 16;
  const size_t rstride = 3 * D_MODEL;
  const bf16* qptr = qkv + ((size_t)b * S_LEN) * rstride + h * HEAD_DIM;
  const bf16* kptr = qptr + D_MODEL;
  const bf16* vptr = qptr + 2 * D_MODEL;

  const int mrow = lane & 15;
  const int khA = (lane < 16) ? 0 : 8;
  const int kbB = (lane < 16) ? 0 : 16;
  const int rhalf = (lane >> 4) << 3;

  // Q A-frags (16 rows x 128 dh = 4 frags of 16x32), kept in registers
  bf16x16 qf[4];
  {
    const bf16* qrow = qptr + (size_t)(qbase + mrow) * rstride;
#pragma unroll
    for (int f = 0; f < 4; f++)
      qf[f] = mk16(ld8(qrow + f * 32 + khA), ld8(qrow + f * 32 + khA + 16));
  }

  floatx8 O[8];
#pragma unroll
  for (int dt = 0; dt < 8; dt++) O[dt] = fzero8();
  float m[8], l[8], alpha[8];
#pragma unroll
  for (int j = 0; j < 8; j++) { m[j] = -1e30f; l[j] = 0.0f; }

  const float scale = 0.08838834764831845f;  // 1/sqrt(128)
  const int kend = q0 + 64;                  // causal bound for this block

  for (int k0 = 0; k0 < kend; k0 += 32) {
    // cooperative stage: K via async DMA, V manually transposed into LDS
#pragma unroll
    for (int i = 0; i < 4; i++) {
      int c = tid + 128 * i;                 // 512 chunks of 8 bf16
      int kr = c >> 4;
      int col8 = (c & 15) * 8;
#if HAVE_ASYNC_LDS
      async_ld_b128(kptr + (size_t)(k0 + kr) * rstride + col8, &Kl[kr][col8]);
#else
      *(bf16x8*)&Kl[kr][col8] = ld8(kptr + (size_t)(k0 + kr) * rstride + col8);
#endif
      bf16x8 vv = ld8(vptr + (size_t)(k0 + kr) * rstride + col8);
#pragma unroll
      for (int e = 0; e < 8; e++) Vt[col8 + e][kr] = vv[e];
    }
#if HAVE_ASYNC_LDS
    wait_async_all();
#endif
    __syncthreads();

    // scores: S[16x32] = Q @ K^T  (two 16x16 C tiles)
    floatx8 s0 = fzero8(), s1 = fzero8();
#pragma unroll
    for (int f = 0; f < 4; f++) {
      int dh = f * 32 + khA;
      bf16x16 kf0 = mk16(ld8(&Kl[mrow][dh]),      ld8(&Kl[mrow][dh + 16]));
      bf16x16 kf1 = mk16(ld8(&Kl[16 + mrow][dh]), ld8(&Kl[16 + mrow][dh + 16]));
      s0 = wmma_bf16(qf[f], kf0, s0);
      s1 = wmma_bf16(qf[f], kf1, s1);
    }

    // online softmax, causal mask, P -> LDS (A-layout staging)
#pragma unroll
    for (int j = 0; j < 8; j++) {
      float a0 = s0[j] * scale;
      float a1 = s1[j] * scale;
      int rowg = qbase + rhalf + j;
      int kc = k0 + (lane & 15);
      if (kc > rowg)      a0 = -1e9f;
      if (kc + 16 > rowg) a1 = -1e9f;
      float mx = fmaxf(a0, a1);
#pragma unroll
      for (int t = 1; t < 16; t <<= 1) mx = fmaxf(mx, __shfl_xor(mx, t, 32));
      float mn = fmaxf(m[j], mx);
      float al = __expf(m[j] - mn);
      float p0 = __expf(a0 - mn);
      float p1 = __expf(a1 - mn);
      float ps = p0 + p1;
#pragma unroll
      for (int t = 1; t < 16; t <<= 1) ps += __shfl_xor(ps, t, 32);
      l[j] = l[j] * al + ps;
      m[j] = mn;
      alpha[j] = al;
      Pl[wave][rhalf + j][lane & 15]        = f2bf(p0);
      Pl[wave][rhalf + j][16 + (lane & 15)] = f2bf(p1);
    }
#pragma unroll
    for (int dt = 0; dt < 8; dt++)
#pragma unroll
      for (int j = 0; j < 8; j++) O[dt][j] *= alpha[j];

    // O += P @ V  (P as A-frag via LDS; V B-frags contiguous from Vt)
    bf16x16 pf = mk16(ld8(&Pl[wave][mrow][khA]), ld8(&Pl[wave][mrow][khA + 16]));
#pragma unroll
    for (int dt = 0; dt < 8; dt++) {
      bf16x16 vf = mk16(ld8(&Vt[dt * 16 + mrow][khA]), ld8(&Vt[dt * 16 + mrow][khA + 16]));
      O[dt] = wmma_bf16(pf, vf, O[dt]);
    }
    __syncthreads();
  }

  // normalize + store (bf16 [B,S,D] with head offset)
#pragma unroll
  for (int j = 0; j < 8; j++) {
    float inv = 1.0f / l[j];
    int rowg = qbase + rhalf + j;
    bf16* orow = out + ((size_t)b * S_LEN + rowg) * D_MODEL + h * HEAD_DIM;
#pragma unroll
    for (int dt = 0; dt < 8; dt++)
      orow[dt * 16 + (lane & 15)] = f2bf(O[dt][j] * inv);
  }
}

// ---------- launch ----------
extern "C" void kernel_launch(void* const* d_in, const int* in_sizes, int n_in,
                              void* d_out, int out_size, void* d_ws, size_t ws_size,
                              hipStream_t stream) {
  const float* x      = (const float*)d_in[0];
  // d_in[1] = attention_mask (causal; applied analytically)
  const float* W_qkv  = (const float*)d_in[2];
  const float* A_qkv  = (const float*)d_in[3];
  const float* B_qkv  = (const float*)d_in[4];
  const float* W_o    = (const float*)d_in[5];
  const float* A_o    = (const float*)d_in[6];
  const float* B_o    = (const float*)d_in[7];
  float* out = (float*)d_out;

  char* ws = (char*)d_ws;
  size_t off = 0;
  bf16* xb    = (bf16*)(ws + off); off += (size_t)M_ROWS * D_MODEL * 2;        // 33.5 MB
  bf16* wqkvb = (bf16*)(ws + off); off += (size_t)3 * D_MODEL * D_MODEL * 2;   // 25.2 MB
  bf16* wob   = (bf16*)(ws + off); off += (size_t)D_MODEL * D_MODEL * 2;       //  8.4 MB
  bf16* qkvb  = (bf16*)(ws + off); off += (size_t)M_ROWS * 3 * D_MODEL * 2;    // 100.7 MB
  bf16* attnb = (bf16*)(ws + off); off += (size_t)M_ROWS * D_MODEL * 2;        // 33.5 MB
  float* pq   = (float*)(ws + off); off += (size_t)M_ROWS * R_LORA * 4;
  float* po   = (float*)(ws + off); off += (size_t)M_ROWS * R_LORA * 4;

  // 1) casts to bf16
  {
    int n4 = (M_ROWS * D_MODEL) / 4;
    cast_f32_bf16_kernel<<<(n4 + 255) / 256, 256, 0, stream>>>(x, xb, n4);
    n4 = (3 * D_MODEL * D_MODEL) / 4;
    cast_f32_bf16_kernel<<<(n4 + 255) / 256, 256, 0, stream>>>(W_qkv, wqkvb, n4);
    n4 = (D_MODEL * D_MODEL) / 4;
    cast_f32_bf16_kernel<<<(n4 + 255) / 256, 256, 0, stream>>>(W_o, wob, n4);
  }
  // 2) LoRA P for qkv (from f32 x for full precision)
  lora_p_kernel<float><<<M_ROWS, 256, 0, stream>>>(x, A_qkv, pq, D_MODEL);
  // 3) qkv = x @ W_qkv^T + P @ B_qkv^T
  gemm_lora_kernel<bf16><<<dim3(3 * D_MODEL / 128, M_ROWS / 128), 256, 0, stream>>>(
      xb, wqkvb, pq, B_qkv, qkvb, 3 * D_MODEL, D_MODEL);
  // 4) causal attention
  attn_kernel<<<dim3(S_LEN / 64, BATCH * N_HEADS), 128, 0, stream>>>(qkvb, attnb);
  // 5) LoRA P for output proj
  lora_p_kernel<bf16><<<M_ROWS, 256, 0, stream>>>(attnb, A_o, po, D_MODEL);
  // 6) out = attn @ W_o^T + P_o @ B_o^T  (f32 output)
  gemm_lora_kernel<float><<<dim3(D_MODEL / 128, M_ROWS / 128), 256, 0, stream>>>(
      attnb, wob, po, B_o, out, D_MODEL, D_MODEL);
}
